// BiLSTM_CRF_71837622993527
// MI455X (gfx1250) — matrix operand order repaired and checked by reading, not compiled, not run on hardware
//
#include <hip/hip_runtime.h>
#include <hip/hip_bf16.h>

typedef _Float16 v16h __attribute__((ext_vector_type(16)));
typedef _Float16 v8h  __attribute__((ext_vector_type(8)));
typedef float    v8f  __attribute__((ext_vector_type(8)));

union HFrag { v16h v; v8h h[2]; };
union H8    { v8h v; _Float16 e[8]; };

#define B_  64
#define L_  512
#define D_  256
#define H_  256
#define G4_ 1024   // 4*H
#define T_  11

__device__ __forceinline__ float sigf(float x) {
    return 1.0f / (1.0f + __expf(-x));
}
__device__ __forceinline__ float tanh_f(float x) {
    // tanh(x) = 2*sigmoid(2x) - 1 (uses v_exp_f32 transcendental)
    return 2.0f * sigf(2.0f * x) - 1.0f;
}

// ---------------- weight conversion (fp32 -> f16) ----------------
extern "C" __global__ void k_cvt_w(const float* __restrict__ wihf, const float* __restrict__ whhf,
                                   const float* __restrict__ wihb, const float* __restrict__ whhb,
                                   _Float16* __restrict__ wih16, _Float16* __restrict__ whh16) {
    int idx = blockIdx.x * 256 + threadIdx.x;          // 0 .. 4*262144-1
    int m = idx >> 18, r = idx & 262143;
    const float* s = (m == 0) ? wihf : (m == 1) ? wihb : (m == 2) ? whhf : whhb;
    _Float16* dpt = (m < 2) ? wih16 : whh16;
    dpt[(m & 1) * 262144 + r] = (_Float16)s[r];
}

extern "C" __global__ void k_pad_wout(const float* __restrict__ wout, _Float16* __restrict__ wout16) {
    int idx = blockIdx.x * 256 + threadIdx.x;          // 0 .. 16*512-1
    int n = idx >> 9, k = idx & 511;
    wout16[idx] = (n < T_) ? (_Float16)wout[n * 512 + k] : (_Float16)0.0f;
}

// ---------------- embedding gather -> f16 ----------------
extern "C" __global__ void k_embed(const int* __restrict__ sent, const float* __restrict__ emb,
                                   _Float16* __restrict__ x16) {
    int idx = blockIdx.x * 256 + threadIdx.x;          // 0 .. 32768*256-1
    int m = idx >> 8, d = idx & 255;
    x16[idx] = (_Float16)emb[(size_t)sent[m] * D_ + d];
}

// ---------------- input projection GEMM: Gt = x @ W_ih^T + b  (both dirs) ----------------
// Per wave: 1 m-tile x 8 n-tiles, K=256 in 8 WMMA steps. A-frag reused across n-tiles.
// Output stored TRANSPOSED as Gt[dir][l][n][b] so the serial recurrent loop can seed
// its accumulators with contiguous 16B vector loads (b is the fastest axis).
extern "C" __global__ void k_in_gemm(const _Float16* __restrict__ x16,
                                     const _Float16* __restrict__ wih16,
                                     const float* __restrict__ bf, const float* __restrict__ bb,
                                     _Float16* __restrict__ G) {
    const int dir = blockIdx.y;
    const _Float16* W = wih16 + dir * (G4_ * D_);      // [N=1024][K=256] row-major == B operand
    const float* bias = dir ? bb : bf;
    _Float16* Gd = G + (size_t)dir * 32768 * G4_;

    int w = threadIdx.x >> 5;
    int lane = threadIdx.x & 31, l15 = lane & 15, hi = lane >> 4;
    int task = blockIdx.x * 8 + w;                     // 16384 wave-tasks per dir
    int mtile = task >> 3, ngroup = task & 7;
    int row = mtile * 16 + l15;                        // row of x: m = b*L + l

    v8f acc[8];
#pragma unroll
    for (int i = 0; i < 8; i++) acc[i] = (v8f)0.0f;

#pragma unroll
    for (int k8 = 0; k8 < 8; k8++) {
        int kb = k8 * 32;
        HFrag a;
        a.h[0] = *(const v8h*)(x16 + (size_t)row * D_ + kb + 8 * hi);
        a.h[1] = *(const v8h*)(x16 + (size_t)row * D_ + kb + 16 + 8 * hi);
#pragma unroll
        for (int nt = 0; nt < 8; nt++) {
            int n0 = (ngroup * 8 + nt) * 16;
            HFrag b;
            const _Float16* wp = W + (size_t)(n0 + l15) * D_ + kb + 8 * hi;
            b.h[0] = *(const v8h*)(wp);
            b.h[1] = *(const v8h*)(wp + 16);
            acc[nt] = __builtin_amdgcn_wmma_f32_16x16x32_f16(
                false, a.v, false, b.v, (short)0, acc[nt], false, false);
        }
    }
#pragma unroll
    for (int nt = 0; nt < 8; nt++) {
        int n = (ngroup * 8 + nt) * 16 + l15;
        float bv = bias[n];
#pragma unroll
        for (int g = 0; g < 8; g++) {
            int m = mtile * 16 + g + 8 * hi;           // m = b*512 + l (tile never crosses b)
            int b = m >> 9, l = m & 511;
            Gd[((size_t)l * G4_ + n) * B_ + b] = (_Float16)(acc[nt][g] + bv);
        }
    }
}

// ---------------- persistent recurrent LSTM: one workgroup per direction ----------------
// 32 waves, each wave owns (mt = w>>3, h-tiles ht0..ht0+1) and ALL 4 gate tiles,
// so i/f/g/o of a given (b,h) land in the same lane -> gate fusion in registers.
// Accumulators seeded from transposed Gt with one contiguous v8h load per fragment;
// next step's seed region is prefetched (global_prefetch) behind the WMMA phase.
extern "C" __global__ void __launch_bounds__(1024) k_lstm(
        const _Float16* __restrict__ whh16, const _Float16* __restrict__ G,
        const float* __restrict__ h0, const float* __restrict__ c0,
        _Float16* __restrict__ hcat) {
    __shared__ _Float16 hbuf[B_ * H_];                 // 32 KB

    const int dir = blockIdx.x;
    const _Float16* W = whh16 + dir * (G4_ * H_);      // [N=1024][K=256]
    const _Float16* Gd = G + (size_t)dir * 32768 * G4_;

    int tid = threadIdx.x;
    int w = tid >> 5, lane = tid & 31, l15 = lane & 15, hi = lane >> 4;
    int mt = w >> 3;                                   // both combos of a wave share mt
    int ht0 = (2 * w) & 15;

    // init h in LDS
#pragma unroll
    for (int i = 0; i < 16; i++) {
        int e = tid + i * 1024;
        hbuf[e] = (_Float16)h0[dir * B_ * H_ + e];
    }
    // init c in registers
    float cst[2][8];
#pragma unroll
    for (int cb = 0; cb < 2; cb++) {
        int h = (ht0 + cb) * 16 + l15;
#pragma unroll
        for (int g = 0; g < 8; g++) {
            int b = mt * 16 + g + 8 * hi;
            cst[cb][g] = c0[dir * B_ * H_ + b * H_ + h];
        }
    }
    __syncthreads();

    for (int t = 0; t < L_; t++) {
        int lt = dir ? (L_ - 1 - t) : t;
        v8f acc[2][4];
        // seed accumulators with x@Wih^T + b from transposed Gt: one 16B load per frag
#pragma unroll
        for (int cb = 0; cb < 2; cb++) {
            int ht = ht0 + cb;
#pragma unroll
            for (int gt = 0; gt < 4; gt++) {
                int n = gt * H_ + ht * 16 + l15;
                H8 s;
                s.v = *(const v8h*)(Gd + ((size_t)lt * G4_ + n) * B_ + mt * 16 + 8 * hi);
#pragma unroll
                for (int g = 0; g < 8; g++) acc[cb][gt][g] = (float)s.e[g];
            }
        }
        // prefetch next step's seed fragments behind the matrix work
        if (t + 1 < L_) {
            int ltn = dir ? (lt - 1) : (lt + 1);
#pragma unroll
            for (int cb = 0; cb < 2; cb++) {
#pragma unroll
                for (int gt = 0; gt < 4; gt++) {
                    int n = gt * H_ + (ht0 + cb) * 16 + l15;
                    __builtin_prefetch(Gd + ((size_t)ltn * G4_ + n) * B_ + mt * 16 + 8 * hi, 0, 1);
                }
            }
        }
        // h @ Whh^T, K=256 in 8 WMMA steps; A from LDS, B streamed from L2
#pragma unroll
        for (int k8 = 0; k8 < 8; k8++) {
            int kb = k8 * 32;
            HFrag a;
            int brow = mt * 16 + l15;
            a.h[0] = *(const v8h*)(&hbuf[brow * H_ + kb + 8 * hi]);
            a.h[1] = *(const v8h*)(&hbuf[brow * H_ + kb + 16 + 8 * hi]);
#pragma unroll
            for (int cb = 0; cb < 2; cb++) {
                int ht = ht0 + cb;
#pragma unroll
                for (int gt = 0; gt < 4; gt++) {
                    int n0 = gt * H_ + ht * 16;
                    HFrag b;
                    const _Float16* wp = W + (size_t)(n0 + l15) * H_ + kb + 8 * hi;
                    b.h[0] = *(const v8h*)(wp);
                    b.h[1] = *(const v8h*)(wp + 16);
                    acc[cb][gt] = __builtin_amdgcn_wmma_f32_16x16x32_f16(
                        false, a.v, false, b.v, (short)0, acc[cb][gt], false, false);
                }
            }
        }
        __syncthreads();   // all hbuf reads done before overwrite
        // gate fusion entirely in-lane
#pragma unroll
        for (int cb = 0; cb < 2; cb++) {
            int h = (ht0 + cb) * 16 + l15;
#pragma unroll
            for (int g = 0; g < 8; g++) {
                int b = mt * 16 + g + 8 * hi;
                float ig = sigf(acc[cb][0][g]);
                float fg = sigf(acc[cb][1][g]);
                float gg = tanh_f(acc[cb][2][g]);
                float og = sigf(acc[cb][3][g]);
                float c = fg * cst[cb][g] + ig * gg;
                cst[cb][g] = c;
                float hn = og * tanh_f(c);
                hbuf[b * H_ + h] = (_Float16)hn;
                hcat[((size_t)b * L_ + lt) * 512 + dir * H_ + h] = (_Float16)hn;
            }
        }
        __syncthreads();   // hbuf ready for next step
    }
}

// ---------------- output projection: logits = hcat @ Wout^T + b_out ----------------
extern "C" __global__ void k_logits(const _Float16* __restrict__ hcat,
                                    const _Float16* __restrict__ wout16,
                                    const float* __restrict__ bout,
                                    float* __restrict__ logits) {
    int w = threadIdx.x >> 5, lane = threadIdx.x & 31, l15 = lane & 15, hi = lane >> 4;
    int mtile = blockIdx.x * 8 + w;
    int row = mtile * 16 + l15;
    v8f acc = (v8f)0.0f;
#pragma unroll
    for (int k16 = 0; k16 < 16; k16++) {
        int kb = k16 * 32;
        HFrag a, b;
        a.h[0] = *(const v8h*)(hcat + (size_t)row * 512 + kb + 8 * hi);
        a.h[1] = *(const v8h*)(hcat + (size_t)row * 512 + kb + 16 + 8 * hi);
        const _Float16* wp = wout16 + l15 * 512 + kb + 8 * hi;
        b.h[0] = *(const v8h*)(wp);
        b.h[1] = *(const v8h*)(wp + 16);
        acc = __builtin_amdgcn_wmma_f32_16x16x32_f16(
            false, a.v, false, b.v, (short)0, acc, false, false);
    }
    if (l15 < T_) {
        float bv = bout[l15];
#pragma unroll
        for (int g = 0; g < 8; g++) {
            int m = mtile * 16 + g + 8 * hi;
            logits[(size_t)m * T_ + l15] = acc[g] + bv;
        }
    }
}

// ---------------- Viterbi (T=11): one thread per batch ----------------
extern "C" __global__ void k_viterbi(const float* __restrict__ logits,
                                     const float* __restrict__ trans,
                                     unsigned char* __restrict__ bp,
                                     float* __restrict__ out) {
    __shared__ float tr[T_ * T_];
    int tid = threadIdx.x;   // 64 threads
    for (int i = tid; i < T_ * T_; i += 64) tr[i] = trans[i];
    __syncthreads();
    int b = tid;
    float prev[T_];
#pragma unroll
    for (int j = 0; j < T_; j++) prev[j] = logits[(size_t)b * L_ * T_ + j];
    for (int t = 1; t < L_; t++) {
        float cur[T_];
#pragma unroll
        for (int j = 0; j < T_; j++) {
            float best = prev[0] + tr[j];
            int bi = 0;
#pragma unroll
            for (int i = 1; i < T_; i++) {
                float v = prev[i] + tr[i * T_ + j];
                if (v > best) { best = v; bi = i; }
            }
            bp[((size_t)b * L_ + t) * T_ + j] = (unsigned char)bi;
            cur[j] = logits[((size_t)b * L_ + t) * T_ + j] + best;
        }
#pragma unroll
        for (int j = 0; j < T_; j++) prev[j] = cur[j];
    }
    float best = prev[0]; int bi = 0;
#pragma unroll
    for (int j = 1; j < T_; j++) if (prev[j] > best) { best = prev[j]; bi = j; }
    out[b] = best;
    int state = bi;
    out[64 + (size_t)b * L_ + (L_ - 1)] = (float)state;
    for (int t = L_ - 1; t >= 1; t--) {
        state = bp[((size_t)b * L_ + t) * T_ + state];
        out[64 + (size_t)b * L_ + (t - 1)] = (float)state;
    }
}

// ---------------- launch ----------------
extern "C" void kernel_launch(void* const* d_in, const int* in_sizes, int n_in,
                              void* d_out, int out_size, void* d_ws, size_t ws_size,
                              hipStream_t stream) {
    (void)in_sizes; (void)n_in; (void)out_size; (void)ws_size;
    const int*   sent  = (const int*)d_in[0];
    const float* emb   = (const float*)d_in[1];
    const float* wihf  = (const float*)d_in[2];
    const float* whhf  = (const float*)d_in[3];
    const float* bf    = (const float*)d_in[4];
    const float* wihb  = (const float*)d_in[5];
    const float* whhb  = (const float*)d_in[6];
    const float* bb    = (const float*)d_in[7];
    const float* wout  = (const float*)d_in[8];
    const float* bout  = (const float*)d_in[9];
    const float* trans = (const float*)d_in[10];
    const float* h0    = (const float*)d_in[11];
    const float* c0    = (const float*)d_in[12];

    char* ws = (char*)d_ws;
    _Float16* x16    = (_Float16*)(ws);                    // 16 MB
    _Float16* wih16  = (_Float16*)(ws + 16777216);         // 1 MB  (f then b)
    _Float16* whh16  = (_Float16*)(ws + 17825792);         // 1 MB
    _Float16* wout16 = (_Float16*)(ws + 18874368);         // 16 KB ([16][512], padded)
    _Float16* G      = (_Float16*)(ws + 18890752);         // 128 MB (2 dirs, [l][n][b])
    _Float16* hcat   = (_Float16*)(ws + 153108480);        // 32 MB ([B][L][512])
    float*    logits = (float*)   (ws + 186662912);        // 1.4 MB
    unsigned char* bp = (unsigned char*)(ws + 188104704);  // 352 KB
    float* out = (float*)d_out;

    k_cvt_w   <<<dim3(4096),     dim3(256),  0, stream>>>(wihf, whhf, wihb, whhb, wih16, whh16);
    k_pad_wout<<<dim3(32),       dim3(256),  0, stream>>>(wout, wout16);
    k_embed   <<<dim3(32768),    dim3(256),  0, stream>>>(sent, emb, x16);
    k_in_gemm <<<dim3(2048, 2),  dim3(256),  0, stream>>>(x16, wih16, bf, bb, G);
    k_lstm    <<<dim3(2),        dim3(1024), 0, stream>>>(whh16, G, h0, c0, hcat);
    k_logits  <<<dim3(256),      dim3(256),  0, stream>>>(hcat, wout16, bout, logits);
    k_viterbi <<<dim3(1),        dim3(64),   0, stream>>>(logits, trans, bp, out);
}
